// MHA_20100446945549
// MI455X (gfx1250) — matrix operand order repaired and checked by reading, not compiled, not run on hardware
//
#include <hip/hip_runtime.h>
#include <hip/hip_bf16.h>

// ---------------------------------------------------------------------------
// MHA forward for MI455X (gfx1250, wave32, WMMA 16x16x32 f16->f32).
//   B=2, S=2048, D=768, H=12, Dh=64
// Pipeline:
//   wtrans      : W[in][out] f32 -> Wt[out][in] f16           (x4 weights)
//   gemm16<0..2>: Q/K/V projections, f16 outputs in attention layouts
//   attn_kernel : flash attention; K/V tiles staged to LDS with
//                 global_load_async_to_lds_b128 (ASYNCcnt), double-buffered,
//                 shared by 8 waves per workgroup
//   gemm16<3>   : output projection -> f32 d_out
// ---------------------------------------------------------------------------

#define DMODEL 768
#define NHEADS 12
#define HDIM   64
#define SEQ    2048
#define BATCH  2
#define MROWS  (BATCH * SEQ)   // 4096
#define QWAVES 8               // waves per attention block

typedef __attribute__((ext_vector_type(16))) _Float16 v16h;
typedef __attribute__((ext_vector_type(8)))  _Float16 v8h;
typedef __attribute__((ext_vector_type(8)))  float    v8f;
typedef __attribute__((ext_vector_type(4)))  int      v4i;

__device__ __forceinline__ v16h join8(v8h a, v8h b) {
    v16h r;
#pragma unroll
    for (int i = 0; i < 8; ++i) { r[i] = a[i]; r[i + 8] = b[i]; }
    return r;
}

__device__ __forceinline__ v8f wmma_f16(v16h a, v16h b, v8f c) {
    // D = A(16x32 f16) * B(32x16 f16) + C(16x16 f32)
    return __builtin_amdgcn_wmma_f32_16x16x32_f16(
        /*neg_a=*/false, a, /*neg_b=*/false, b,
        /*c_mod=*/(short)0, c, /*reuse_a=*/false, /*reuse_b=*/false);
}

__device__ __forceinline__ int xor16_i(int v) {
    int lane = (int)(threadIdx.x & 31u);
    return __builtin_amdgcn_ds_bpermute(((lane ^ 16) << 2), v);
}
__device__ __forceinline__ float xor16_f(float v) {
    return __int_as_float(xor16_i(__float_as_int(v)));
}

// Async copy 16B/lane from global to LDS (ASYNCcnt-tracked DMA path).
// VFLAT async form: global_load_async_to_lds_b128 vdst(LDS addr), vaddr64, off
__device__ __forceinline__ void async_b128(const _Float16* g, const _Float16* l) {
    unsigned loff = (unsigned)(uintptr_t)(const void*)l;  // low 32b = LDS offset
    asm volatile("global_load_async_to_lds_b128 %0, %1, off"
                 :: "v"(loff), "v"(g)
                 : "memory");
}

// ---------------------------------------------------------------------------
// Weight transpose + f32->f16 convert: Wt[n][k] = (f16) W[k][n]
// grid (12, 192), block 256: 64 k-lanes x 4 n-rows per block.
// ---------------------------------------------------------------------------
__global__ void wtrans(const float* __restrict__ W, _Float16* __restrict__ Wt) {
    int k = blockIdx.x * 64 + (threadIdx.x & 63);
    int n = blockIdx.y * 4 + (threadIdx.x >> 6);
    Wt[(size_t)n * DMODEL + k] = (_Float16)W[(size_t)k * DMODEL + n];
}

// ---------------------------------------------------------------------------
// GEMM: C[4096 x 768] = A[4096 x 768] @ Wt^T + bias, via WMMA.
// Wave computes a 16x64 tile; K-loop step 32.
// MODE 0: A=f32, out -> Q f16 [bh][s][64] scaled by 1/8
// MODE 1: A=f32, out -> K f16 [bh][s][64]
// MODE 2: A=f32, out -> V f16 transposed [bh][64][S]
// MODE 3: A=f16 (attention output [b][s][768]), out -> f32 [4096][768]
// grid (12, 64), block 128 (4 waves).
// ---------------------------------------------------------------------------
template <int MODE>
__global__ __launch_bounds__(128) void gemm16(
    const void* __restrict__ Ap,
    const _Float16* __restrict__ Wt,   // [768][768], n-major
    const float* __restrict__ bias,    // [768]
    void* __restrict__ Out)
{
    const int lane = (int)(threadIdx.x & 31u);
    const int wave = (int)(threadIdx.x >> 5);
    const int l16  = lane & 15;
    const bool hi  = lane >= 16;
    const int o    = hi ? 8 : 0;

    const int nb = blockIdx.x * 64;
    const int mb = (blockIdx.y * 4 + wave) * 16;

    v8f zero = {};
    v8f acc[4];
#pragma unroll
    for (int t = 0; t < 4; ++t) acc[t] = zero;

    const int arow = mb + l16;   // A fragment: lanes l and l+16 share row l16

    for (int kb = 0; kb < DMODEL; kb += 32) {
        v16h af;
        if (MODE == 3) {
            const _Float16* ar = (const _Float16*)Ap + (size_t)arow * DMODEL + kb;
            af = join8(*(const v8h*)(ar + o), *(const v8h*)(ar + 16 + o));
        } else {
            const float* ar = (const float*)Ap + (size_t)arow * DMODEL + kb;
            v8f f0 = *(const v8f*)(ar + o);
            v8f f1 = *(const v8f*)(ar + 16 + o);
            v8h h0, h1;
#pragma unroll
            for (int i = 0; i < 8; ++i) { h0[i] = (_Float16)f0[i]; h1[i] = (_Float16)f1[i]; }
            af = join8(h0, h1);
        }
#pragma unroll
        for (int t = 0; t < 4; ++t) {
            // B fragment: lane<16 -> col n=nb+16t+l16, rows kb+0..7, kb+16..23;
            //             lane>=16 -> same col set, rows kb+8..15, kb+24..31.
            const _Float16* wr =
                Wt + (size_t)(nb + t * 16 + l16) * DMODEL + kb;
            v16h bf = join8(*(const v8h*)(wr + o), *(const v8h*)(wr + 16 + o));
            acc[t] = wmma_f16(af, bf, acc[t]);
        }
    }

    // Epilogue. D layout: lane column n = nb+16t+l16; VGPR r row m = mb+r+8*hi.
#pragma unroll
    for (int t = 0; t < 4; ++t) {
        const int n = nb + t * 16 + l16;
        const float bn = bias[n];
        if (MODE == 0 || MODE == 1) {
            const int h = n >> 6, dh = n & 63;
            const float sc = (MODE == 0) ? 0.125f : 1.0f;  // fold 1/sqrt(64) into Q
#pragma unroll
            for (int r = 0; r < 8; ++r) {
                const int m = mb + r + (hi ? 8 : 0);
                const int bidx = m >> 11, s = m & (SEQ - 1);
                _Float16* dst = (_Float16*)Out +
                    (((size_t)(bidx * NHEADS + h) * SEQ + s) * HDIM + dh);
                *dst = (_Float16)((acc[t][r] + bn) * sc);
            }
        } else if (MODE == 2) {
            const int h = n >> 6, dh = n & 63;
            const int m0 = mb + (hi ? 8 : 0);
            const int bidx = m0 >> 11, s0 = m0 & (SEQ - 1);
            v8h pk;
#pragma unroll
            for (int r = 0; r < 8; ++r) pk[r] = (_Float16)(acc[t][r] + bn);
            *(v8h*)((_Float16*)Out +
                    ((size_t)(bidx * NHEADS + h) * HDIM + dh) * SEQ + s0) = pk;
        } else {
#pragma unroll
            for (int r = 0; r < 8; ++r) {
                const int m = mb + r + (hi ? 8 : 0);
                ((float*)Out)[(size_t)m * DMODEL + n] = acc[t][r] + bn;
            }
        }
    }
}

// ---------------------------------------------------------------------------
// Flash attention with async-LDS K/V staging.
// Block: 256 threads (8 waves), one (b,h), 128 consecutive queries.
// Per 32-key block: 8 KB (K tile 32x64 + V^T tile 64x32, f16) DMA'd into LDS
// via global_load_async_to_lds_b128, double-buffered; waves compute
//   scoresT = K(16x64) @ Q^T(64x16), outT += V^T(16x32) @ P^T(32x16)
// with fragments read from LDS. grid 384.
// ---------------------------------------------------------------------------
__global__ __launch_bounds__(QWAVES * 32) void attn_kernel(
    const _Float16* __restrict__ Q,    // [BH][S][64], prescaled by 1/8
    const _Float16* __restrict__ K,    // [BH][S][64]
    const _Float16* __restrict__ Vt,   // [BH][64][S]
    _Float16* __restrict__ AO)         // [B][S][768]
{
    __shared__ _Float16 sK[2][32 * HDIM];   // [key_local][dh]
    __shared__ _Float16 sV[2][HDIM * 32];   // [dh][key_local]

    const int tid  = (int)threadIdx.x;
    const int lane = tid & 31;
    const int wave = tid >> 5;
    const int bh   = blockIdx.x >> 4;                       // 16 blocks per head
    const int qb   = ((blockIdx.x & 15) * QWAVES + wave) * 16;
    const int l16  = lane & 15;
    const bool hi  = lane >= 16;
    const int o    = hi ? 8 : 0;

    const _Float16* Kb = K  + (size_t)bh * SEQ * HDIM;
    const _Float16* Vb = Vt + (size_t)bh * HDIM * SEQ;

    // Q^T B-fragments (reused across all key blocks): col = query l16,
    // K-rows = dh chunk; upper half-lanes take dh+8 (8-wide slices).
    const _Float16* qr = Q + ((size_t)bh * SEQ + qb + l16) * HDIM;
    const v16h qB0 = join8(*(const v8h*)(qr + o), *(const v8h*)(qr + 16 + o));        // dh 0..31
    const v16h qB1 = join8(*(const v8h*)(qr + 32 + o), *(const v8h*)(qr + 48 + o));   // dh 32..63

    v8f zero = {};
    v8f acc[4];
#pragma unroll
    for (int t = 0; t < 4; ++t) acc[t] = zero;
    float m_run = -1e30f, l_run = 0.0f;

    // Each of 256 threads DMAs one 16B chunk of the K tile and one of the V
    // tile per key block (2 async instructions per wave per block).
    auto issue = [&](int kb, int buf) {
        const int c = tid;                              // 0..255
        const _Float16* gk = Kb + (size_t)(kb + (c >> 3)) * HDIM + (c & 7) * 8;
        async_b128(gk, &sK[buf][c * 8]);
        const _Float16* gv = Vb + (size_t)(c >> 2) * SEQ + kb + (c & 3) * 8;
        async_b128(gv, &sV[buf][c * 8]);
    };

    issue(0, 0);
    for (int kb = 0; kb < SEQ; kb += 32) {
        const int cur = (kb >> 5) & 1;
        if (kb + 32 < SEQ) {
            issue(kb + 32, cur ^ 1);
            // async loads complete in order: <=2 outstanding => current tile done
            asm volatile("s_wait_asynccnt 0x2" ::: "memory");
        } else {
            asm volatile("s_wait_asynccnt 0x0" ::: "memory");
        }
        __syncthreads();

        const _Float16* sk = &sK[cur][0];
        const _Float16* sv = &sV[cur][0];

        // A fragments from K tile (tile0: keys 0..15, tile1: keys 16..31)
        const _Float16* kr0 = sk + (size_t)l16 * HDIM;
        const _Float16* kr1 = sk + (size_t)(16 + l16) * HDIM;
        v16h a00 = join8(*(const v8h*)(kr0 + o),      *(const v8h*)(kr0 + 16 + o));
        v16h a01 = join8(*(const v8h*)(kr0 + 32 + o), *(const v8h*)(kr0 + 48 + o));
        v16h a10 = join8(*(const v8h*)(kr1 + o),      *(const v8h*)(kr1 + 16 + o));
        v16h a11 = join8(*(const v8h*)(kr1 + 32 + o), *(const v8h*)(kr1 + 48 + o));

        v8f s0 = zero, s1 = zero;
        s0 = wmma_f16(a00, qB0, s0);
        s0 = wmma_f16(a01, qB1, s0);
        s1 = wmma_f16(a10, qB0, s1);
        s1 = wmma_f16(a11, qB1, s1);

        // Online softmax. Lane holds query col l16; its 16 scores cover half
        // the 32-key block; partner lane (xor 16) holds the other half.
        float mx = -1e30f;
#pragma unroll
        for (int r = 0; r < 8; ++r) { mx = fmaxf(mx, s0[r]); mx = fmaxf(mx, s1[r]); }
        mx = fmaxf(mx, xor16_f(mx));
        const float m_new = fmaxf(m_run, mx);
        const float alpha = __expf(m_run - m_new);

        float e0[8], e1[8], ls = 0.0f;
#pragma unroll
        for (int r = 0; r < 8; ++r) {
            e0[r] = __expf(s0[r] - m_new);
            e1[r] = __expf(s1[r] - m_new);
            ls += e0[r] + e1[r];
        }
        ls += xor16_f(ls);
        l_run = l_run * alpha + ls;
        m_run = m_new;
#pragma unroll
        for (int t = 0; t < 4; ++t)
#pragma unroll
            for (int r = 0; r < 8; ++r) acc[t][r] *= alpha;

        // Build P^T B-fragment: lane<16 needs keys 0..15 (own tile0 rows 0-7
        // + partner tile0 rows 8-15); lane>=16 needs keys 16..31 (partner
        // tile1 rows 0-7 + own tile1 rows 8-15). Exchange via ds_bpermute ^16.
        v8h own0, own1;
#pragma unroll
        for (int r = 0; r < 8; ++r) { own0[r] = (_Float16)e0[r]; own1[r] = (_Float16)e1[r]; }
        v4i sp = __builtin_bit_cast(v4i, hi ? own0 : own1);
        v4i rp;
#pragma unroll
        for (int i = 0; i < 4; ++i) rp[i] = xor16_i(sp[i]);
        v8h recv = __builtin_bit_cast(v8h, rp);
        v8h lowh  = hi ? recv : own0;
        v8h highh = hi ? own1 : recv;
        v16h pB = join8(lowh, highh);

        // outT accumulate: A = V^T dh-tile from LDS (row dh = 16t+l16,
        // local key stride 32; key slices split across half-lanes).
#pragma unroll
        for (int t = 0; t < 4; ++t) {
            const _Float16* vr = sv + (size_t)(t * 16 + l16) * 32;
            v16h va = *(const v16h*)(vr + (hi ? 16 : 0));
            acc[t] = wmma_f16(va, pB, acc[t]);
        }

        __syncthreads();   // all waves done with buf `cur` before it is refilled
    }

    // Normalize and store: lane's query col q=l16; dh = 16t + r + 8*hi.
    const float inv = 1.0f / l_run;
    const int b = bh / NHEADS, h = bh % NHEADS;
    const int row = qb + l16;
    _Float16* out = AO + ((size_t)(b * SEQ + row)) * DMODEL + h * HDIM;
#pragma unroll
    for (int t = 0; t < 4; ++t) {
        v8h pk;
#pragma unroll
        for (int r = 0; r < 8; ++r) pk[r] = (_Float16)(acc[t][r] * inv);
        *(v8h*)(out + t * 16 + (hi ? 8 : 0)) = pk;
    }
}

// ---------------------------------------------------------------------------
// Host launcher
// ---------------------------------------------------------------------------
extern "C" void kernel_launch(void* const* d_in, const int* in_sizes, int n_in,
                              void* d_out, int out_size, void* d_ws, size_t ws_size,
                              hipStream_t stream) {
    const float* q  = (const float*)d_in[0];
    const float* k  = (const float*)d_in[1];
    const float* v  = (const float*)d_in[2];
    const float* Wq = (const float*)d_in[3];
    const float* bq = (const float*)d_in[4];
    const float* Wk = (const float*)d_in[5];
    const float* bk = (const float*)d_in[6];
    const float* Wv = (const float*)d_in[7];
    const float* bv = (const float*)d_in[8];
    const float* Wo = (const float*)d_in[9];
    const float* bo = (const float*)d_in[10];

    static constexpr size_t WT_SZ  = (size_t)DMODEL * DMODEL * sizeof(_Float16); // 1.125 MB
    static constexpr size_t HDT_SZ = (size_t)BATCH * NHEADS * SEQ * HDIM * sizeof(_Float16); // 6 MB

    char* ws = (char*)d_ws;
    _Float16* WqT = (_Float16*)(ws + 0 * WT_SZ);
    _Float16* WkT = (_Float16*)(ws + 1 * WT_SZ);
    _Float16* WvT = (_Float16*)(ws + 2 * WT_SZ);
    _Float16* WoT = (_Float16*)(ws + 3 * WT_SZ);
    _Float16* Qh  = (_Float16*)(ws + 4 * WT_SZ);
    _Float16* Kh  = (_Float16*)(ws + 4 * WT_SZ + 1 * HDT_SZ);
    _Float16* Vth = (_Float16*)(ws + 4 * WT_SZ + 2 * HDT_SZ);
    _Float16* AO  = (_Float16*)(ws + 4 * WT_SZ + 3 * HDT_SZ);

    dim3 tb(256), tg(DMODEL / 64, DMODEL / 4);
    wtrans<<<tg, tb, 0, stream>>>(Wq, WqT);
    wtrans<<<tg, tb, 0, stream>>>(Wk, WkT);
    wtrans<<<tg, tb, 0, stream>>>(Wv, WvT);
    wtrans<<<tg, tb, 0, stream>>>(Wo, WoT);

    dim3 gb(128), gg(DMODEL / 64, MROWS / 64);
    gemm16<0><<<gg, gb, 0, stream>>>(q, WqT, bq, Qh);
    gemm16<1><<<gg, gb, 0, stream>>>(k, WkT, bk, Kh);
    gemm16<2><<<gg, gb, 0, stream>>>(v, WvT, bv, Vth);

    // 24 heads x 16 blocks, 8 waves x 16 queries per block
    attn_kernel<<<dim3(BATCH * NHEADS * (SEQ / (16 * QWAVES))), dim3(QWAVES * 32), 0, stream>>>(
        Qh, Kh, Vth, AO);

    gemm16<3><<<gg, gb, 0, stream>>>(AO, WoT, bo, d_out);
}
